// AGATCellWithMLP_53523882442786
// MI455X (gfx1250) — matrix operand order repaired and checked by reading, not compiled, not run on hardware
//
#include <hip/hip_runtime.h>

// ---------------- problem constants ----------------
#define B_   8
#define N_   1024
#define Dh   64
#define H_   4
#define QVd  32
#define C_   129
#define CP   160          // C padded to multiple of 32 (bias channel at 129)
#define BN   (B_*N_)      // 8192
#define M_   4096
#define CATW 576          // 4*136 head blocks + bias col 544 + pad
#define HIDW 160
#define HKW  5120         // hypernet K: 32 d-blocks * 160
#define NEGV (-9.0e15f)

typedef __attribute__((ext_vector_type(16))) _Float16 v16h;
typedef __attribute__((ext_vector_type(8)))  _Float16 h8;
typedef __attribute__((ext_vector_type(8)))  float    v8f;

// ---------------- WMMA helpers (gfx1250 wave32) ----------------
__device__ __forceinline__ v8f wmma16(v16h a, v16h b, v8f c) {
  // D = A(16x32 f16) * B(32x16 f16) + C(16x16 f32)
  return __builtin_amdgcn_wmma_f32_16x16x32_f16(false, a, false, b, (short)0, c,
                                                false, false);
}
__device__ __forceinline__ v16h cat16(h8 lo, h8 hi) {
  v16h r;
#pragma unroll
  for (int j = 0; j < 8; ++j) { r[j] = lo[j]; r[j + 8] = hi[j]; }
  return r;
}
// A operand: row-major f16 [16 rows, K>=32], row stride multiple of 8.
// lanes 0-15: K 0..15; lanes 16-31: K 16..31.
__device__ __forceinline__ v16h load_a16(const _Float16* base, int stride, int lane) {
  const _Float16* p = base + (lane & 15) * stride + ((lane >> 4) << 4);
  return cat16(*(const h8*)p, *(const h8*)(p + 8));
}
// B operand: stored K-contiguous per column, column stride multiple of 8.
// lanes 0-15: K 0..15 of column lane; lanes 16-31: K 16..31 of column lane-16.
__device__ __forceinline__ v16h load_b16(const _Float16* base, int stride, int lane) {
  const _Float16* p = base + (lane & 15) * stride + ((lane >> 4) << 4);
  return cat16(*(const h8*)p, *(const h8*)(p + 8));
}

// ================= elementwise / prep kernels =================

// combined = [x | h | 1 | 0-pad] ; f32 (dense 129) and f16 (padded 160)
__global__ void k_combined(const float* __restrict__ x, const float* __restrict__ h,
                           float* __restrict__ cf32, _Float16* __restrict__ cf16) {
  int idx = blockIdx.x * blockDim.x + threadIdx.x;
  if (idx >= BN * CP) return;
  int row = idx / CP, c = idx - row * CP;
  float v;
  if (c < 65)       v = x[row * 65 + c];
  else if (c < 129) v = h[row * 64 + (c - 65)];
  else if (c == 129) v = 1.0f;
  else              v = 0.0f;
  cf16[idx] = (_Float16)v;
  if (c < 129) cf32[row * 129 + c] = v;
}

// Wq/Wk [H,129,16] + bias -> transposed padded f16 [H][16][160]
__global__ void k_prep_qk(const float* __restrict__ W, const float* __restrict__ bias,
                          _Float16* __restrict__ Wt) {
  int idx = blockIdx.x * blockDim.x + threadIdx.x;
  if (idx >= H_ * 16 * CP) return;
  int hh = idx / (16 * CP); int r = idx - hh * 16 * CP;
  int kk = r / CP; int c = r - kk * CP;
  float v = 0.f;
  if (c < 129)       v = W[(hh * 129 + c) * 16 + kk];
  else if (c == 129) v = bias[hh * 16 + kk];
  Wt[idx] = (_Float16)v;
}

// Wv [H,129,129] + bv -> f16 [H][144][160]  (col d, K c)
__global__ void k_prep_v(const float* __restrict__ W, const float* __restrict__ bias,
                         _Float16* __restrict__ Wt) {
  int idx = blockIdx.x * blockDim.x + threadIdx.x;
  if (idx >= H_ * 144 * CP) return;
  int hh = idx / (144 * CP); int r = idx - hh * 144 * CP;
  int d = r / CP; int c = r - d * CP;
  float v = 0.f;
  if (d < 129) {
    if (c < 129)       v = W[(hh * 129 + c) * 129 + d];
    else if (c == 129) v = bias[hh * 129 + d];
  }
  Wt[idx] = (_Float16)v;
}

// W1 [516,129] + b1 -> f16 [144][576]; head block h occupies K rows h*136..h*136+128; bias row 544
__global__ void k_prep_w1(const float* __restrict__ W, const float* __restrict__ bias,
                          _Float16* __restrict__ Wt) {
  int idx = blockIdx.x * blockDim.x + threadIdx.x;
  if (idx >= 144 * CATW) return;
  int o = idx / CATW; int k = idx - o * CATW;
  float v = 0.f;
  if (o < 129) {
    if (k < 544) { int hh = k / 136, c = k - hh * 136; if (c < 129) v = W[(hh * 129 + c) * 129 + o]; }
    else if (k == 544) v = bias[o];
  }
  Wt[idx] = (_Float16)v;
}

// W2 [129,129] + b2 -> f16 [144][160]
__global__ void k_prep_w2(const float* __restrict__ W, const float* __restrict__ bias,
                          _Float16* __restrict__ Wt) {
  int idx = blockIdx.x * blockDim.x + threadIdx.x;
  if (idx >= 144 * HIDW) return;
  int o = idx / HIDW; int c = idx - o * HIDW;
  float v = 0.f;
  if (o < 129) {
    if (c < 129)       v = W[c * 129 + o];
    else if (c == 129) v = bias[o];
  }
  Wt[idx] = (_Float16)v;
}

// hypernet W [32,129,64] + b [32,64] -> f16 [64][5120] (d block = 160 K, bias at i=129)
__global__ void k_prep_hg(const float* __restrict__ W, const float* __restrict__ bias,
                          _Float16* __restrict__ Wt) {
  int idx = blockIdx.x * blockDim.x + threadIdx.x;
  if (idx >= 64 * HKW) return;
  int o = idx / HKW; int k = idx - o * HKW;
  int d = k / CP; int i = k - d * CP;
  float v = 0.f;
  if (i < 129)       v = W[(d * 129 + i) * 64 + o];
  else if (i == 129) v = bias[d * 64 + o];
  Wt[idx] = (_Float16)v;
}

// fill f16 buffer with 0, a 1.0 at one column per row
__global__ void k_fill(_Float16* __restrict__ p, int total, int width, int one_col) {
  int idx = blockIdx.x * blockDim.x + threadIdx.x;
  if (idx >= total) return;
  int c = idx % width;
  p[idx] = (c == one_col) ? (_Float16)1.0f : (_Float16)0.0f;
}

// ================= WMMA GEMM kernels (1 wave / output tile set) =================

// q,k projection: [8192 x 160] @ [160 x 16] per head; outputs f16 [H][8192][32] zero-padded K
__global__ void k_qk(const _Float16* __restrict__ cf16, const _Float16* __restrict__ Wqt,
                     const _Float16* __restrict__ Wkt, _Float16* __restrict__ qo,
                     _Float16* __restrict__ ko) {
  int wid = blockIdx.x;                // H_ * 512
  int hh = wid / 512, rt = wid - hh * 512;
  int lane = threadIdx.x;
  const _Float16* A0 = cf16 + (size_t)(rt * 16) * CP;
  const _Float16* Bq = Wqt + hh * 16 * CP;
  const _Float16* Bk = Wkt + hh * 16 * CP;
  v8f aq = {}, ak = {};
#pragma unroll
  for (int kc = 0; kc < 5; ++kc) {
    v16h a  = load_a16(A0 + kc * 32, CP, lane);
    v16h bq = load_b16(Bq + kc * 32, CP, lane);
    v16h bk = load_b16(Bk + kc * 32, CP, lane);
    aq = wmma16(a, bq, aq);
    ak = wmma16(a, bk, ak);
  }
  int col = lane & 15, roff = (lane >> 4) << 3;
#pragma unroll
  for (int r = 0; r < 8; ++r) {
    int row = rt * 16 + roff + r;
    size_t base = ((size_t)hh * BN + row) * 32;
    qo[base + col] = (_Float16)aq[r];  qo[base + 16 + col] = (_Float16)0.f;
    ko[base + col] = (_Float16)ak[r];  ko[base + 16 + col] = (_Float16)0.f;
  }
}

// v projection, stored transposed: v_t [H][B][144][1024]
__global__ void k_v(const _Float16* __restrict__ cf16, const _Float16* __restrict__ Wvt,
                    _Float16* __restrict__ vt) {
  int wid = blockIdx.x;                // H_ * 512 * 9
  int hh = wid / (512 * 9); int rem = wid - hh * 512 * 9;
  int rt = rem / 9, dt = rem - rt * 9;
  int lane = threadIdx.x;
  const _Float16* A0 = cf16 + (size_t)(rt * 16) * CP;
  const _Float16* Bv = Wvt + ((size_t)hh * 144 + dt * 16) * CP;
  v8f acc = {};
#pragma unroll
  for (int kc = 0; kc < 5; ++kc) {
    v16h a = load_a16(A0 + kc * 32, CP, lane);
    v16h b = load_b16(Bv + kc * 32, CP, lane);
    acc = wmma16(a, b, acc);
  }
  int col = lane & 15, roff = (lane >> 4) << 3;
  int d = dt * 16 + col;
  int b_ = (rt * 16) >> 10, n0 = (rt * 16) & 1023;
  h8 outv;
#pragma unroll
  for (int r = 0; r < 8; ++r) outv[r] = (_Float16)acc[r];
  *(h8*)(vt + (((size_t)hh * B_ + b_) * 144 + d) * N_ + n0 + roff) = outv;
}

// scores = leakyrelu(q k^T / 4), adj mask -> f32 [H][B][1024][1024]
__global__ void k_scores(const _Float16* __restrict__ qf, const _Float16* __restrict__ kf,
                         const int* __restrict__ adj, float* __restrict__ sc) {
  int wid = blockIdx.x;                // H_*B_*64*64
  int mt = wid & 63, nt = (wid >> 6) & 63, b_ = (wid >> 12) & 7, hh = wid >> 15;
  int lane = threadIdx.x;
  const _Float16* A = qf + ((size_t)hh * BN + b_ * N_ + nt * 16) * 32;
  const _Float16* Bk = kf + ((size_t)hh * BN + b_ * N_ + mt * 16) * 32;
  v8f acc = {};
  v16h a = load_a16(A, 32, lane);
  v16h b = load_b16(Bk, 32, lane);
  acc = wmma16(a, b, acc);
  int col = lane & 15, roff = (lane >> 4) << 3;
  int m = mt * 16 + col;
#pragma unroll
  for (int r = 0; r < 8; ++r) {
    int n = nt * 16 + roff + r;
    float s = acc[r] * 0.25f;                // 1/sqrt(16)
    s = (s > 0.f) ? s : 0.2f * s;            // leaky relu
    if (adj[n * N_ + m] == 0) s = NEGV;
    sc[(((size_t)hh * B_ + b_) * N_ + n) * N_ + m] = s;
  }
}

// row softmax: one wave per (h,b,n) row of 1024; output f16 attn
__global__ void k_softmax(const float* __restrict__ sc, _Float16* __restrict__ at) {
  size_t row = blockIdx.x;
  int lane = threadIdx.x;
  const float* s = sc + row * N_;
  float mx = -3.0e38f;
#pragma unroll
  for (int j = 0; j < 32; ++j) mx = fmaxf(mx, s[lane + j * 32]);
#pragma unroll
  for (int off = 16; off > 0; off >>= 1) mx = fmaxf(mx, __shfl_xor(mx, off, 32));
  float ev[32]; float sum = 0.f;
#pragma unroll
  for (int j = 0; j < 32; ++j) { ev[j] = __expf(s[lane + j * 32] - mx); sum += ev[j]; }
#pragma unroll
  for (int off = 16; off > 0; off >>= 1) sum += __shfl_xor(sum, off, 32);
  float inv = 1.f / sum;
  _Float16* a = at + row * N_;
#pragma unroll
  for (int j = 0; j < 32; ++j) a[lane + j * 32] = (_Float16)(ev[j] * inv);
}

// hp = attn @ v; one wave computes ALL 9 d-tiles for a 16-row block so the attn
// A-operand (the dominant HBM stream) is read exactly once. v_t stays L2-resident.
__global__ void k_av(const _Float16* __restrict__ at, const _Float16* __restrict__ vt,
                     _Float16* __restrict__ cat) {
  int wid = blockIdx.x;                // H_*B_*64
  int nt = wid & 63; int rem = wid >> 6;
  int b_ = rem & 7; int hh = rem >> 3;
  int lane = threadIdx.x;
  const _Float16* A0 = at + (((size_t)hh * B_ + b_) * N_ + nt * 16) * N_;
  const _Float16* B0 = vt + (((size_t)hh * B_ + b_) * 144) * N_;
  v8f acc[9] = {};
  for (int kc = 0; kc < 32; ++kc) {
    __builtin_prefetch(A0 + (kc + 2) * 32 + (lane & 15) * N_, 0, 0);
    v16h a = load_a16(A0 + kc * 32, N_, lane);
#pragma unroll
    for (int t = 0; t < 9; ++t) {
      v16h b = load_b16(B0 + ((size_t)t * 16) * N_ + kc * 32, N_, lane);
      acc[t] = wmma16(a, b, acc[t]);
    }
  }
  int col = lane & 15, roff = (lane >> 4) << 3;
#pragma unroll
  for (int t = 0; t < 9; ++t) {
    int d = t * 16 + col;
    if (d < 129) {
#pragma unroll
      for (int r = 0; r < 8; ++r) {
        int row = b_ * N_ + nt * 16 + roff + r;
        cat[(size_t)row * CATW + hh * 136 + d] = (_Float16)acc[t][r];
      }
    }
  }
}

// hidden = relu(attn_cat @ W1 + b1), f16 [8192][160] (bias col preset by k_fill)
__global__ void k_mlp1(const _Float16* __restrict__ cat, const _Float16* __restrict__ W1t,
                       _Float16* __restrict__ hid) {
  int wid = blockIdx.x;                // 512*9
  int ot = wid % 9, rt = wid / 9;
  int lane = threadIdx.x;
  const _Float16* A0 = cat + (size_t)(rt * 16) * CATW;
  const _Float16* B0 = W1t + (size_t)(ot * 16) * CATW;
  v8f acc = {};
  for (int kc = 0; kc < 18; ++kc) {
    v16h a = load_a16(A0 + kc * 32, CATW, lane);
    v16h b = load_b16(B0 + kc * 32, CATW, lane);
    acc = wmma16(a, b, acc);
  }
  int col = lane & 15, roff = (lane >> 4) << 3;
  int o = ot * 16 + col;
  if (o < 129) {
#pragma unroll
    for (int r = 0; r < 8; ++r)
      hid[(size_t)(rt * 16 + roff + r) * HIDW + o] = (_Float16)fmaxf(acc[r], 0.f);
  }
}

// combined2 = hidden @ W2 + b2 + combined (skip), f32 [8192][129]
__global__ void k_mlp2(const _Float16* __restrict__ hid, const _Float16* __restrict__ W2t,
                       const float* __restrict__ cf32, float* __restrict__ c2) {
  int wid = blockIdx.x;                // 512*9
  int ot = wid % 9, rt = wid / 9;
  int lane = threadIdx.x;
  const _Float16* A0 = hid + (size_t)(rt * 16) * HIDW;
  const _Float16* B0 = W2t + (size_t)(ot * 16) * HIDW;
  v8f acc = {};
#pragma unroll
  for (int kc = 0; kc < 5; ++kc) {
    v16h a = load_a16(A0 + kc * 32, HIDW, lane);
    v16h b = load_b16(B0 + kc * 32, HIDW, lane);
    acc = wmma16(a, b, acc);
  }
  int col = lane & 15, roff = (lane >> 4) << 3;
  int o = ot * 16 + col;
  if (o < 129) {
#pragma unroll
    for (int r = 0; r < 8; ++r) {
      size_t row = (size_t)(rt * 16 + roff + r);
      c2[row * 129 + o] = acc[r] + cf32[row * 129 + o];
    }
  }
}

// sel gather: f32 [4096][160]: [combined2_row | 1 | 0]
__global__ void k_sel(const float* __restrict__ c2, const int* __restrict__ nb,
                      const int* __restrict__ nn, float* __restrict__ sel) {
  int idx = blockIdx.x * blockDim.x + threadIdx.x;
  if (idx >= M_ * CP) return;
  int m = idx / CP, i = idx - m * CP;
  float v = 0.f;
  if (i < 129) {
    size_t row = (size_t)nb[m] * N_ + nn[m];
    v = c2[row * 129 + i];
  } else if (i == 129) v = 1.f;
  sel[idx] = v;
}

// selc gather: [x_sel | r*h_sel | 1 | 0]
__global__ void k_selc(const float* __restrict__ x, const float* __restrict__ h,
                       const float* __restrict__ rg, const int* __restrict__ nb,
                       const int* __restrict__ nn, float* __restrict__ selc) {
  int idx = blockIdx.x * blockDim.x + threadIdx.x;
  if (idx >= M_ * CP) return;
  int m = idx / CP, i = idx - m * CP;
  float v = 0.f;
  size_t row = (size_t)nb[m] * N_ + nn[m];
  if (i < 65)       v = x[row * 65 + i];
  else if (i < 129) { int o = i - 65; v = rg[m * 64 + o] * h[row * 64 + o]; }
  else if (i == 129) v = 1.f;
  selc[idx] = v;
}

// hypernet GEMM: out[m,o] = sum_{d,i} qv[m,d]*sel[m,i]*W[d,i,o] (+bias via i=129)
// A operand (qv_d * sel_i) built on the fly, shared across ALL 4 output col tiles
// (4 accumulators) so the expensive VALU A-build is amortized 4x.
// mode 0: out = sigmoid(val); mode 1: out = (1-u)*(r*h) + u*tanh(val)
__global__ void k_hyper(const float* __restrict__ selbuf, const float* __restrict__ qv,
                        const _Float16* __restrict__ Wt, float* __restrict__ outbuf,
                        int mode, const float* __restrict__ rg, const float* __restrict__ ug,
                        const float* __restrict__ hin, const int* __restrict__ nb,
                        const int* __restrict__ nn) {
  int rt = blockIdx.x;                 // 256 row tiles
  int lane = threadIdx.x;
  int arow = rt * 16 + (lane & 15);
  int kba = (lane >> 4) << 4;
  const float* selrow = selbuf + (size_t)arow * CP;
  const float* qvrow = qv + (size_t)arow * QVd;
  const _Float16* Bcol = Wt + (size_t)(lane & 15) * HKW + ((lane >> 4) << 4);
  v8f acc[4] = {};
  for (int kc = 0; kc < 160; ++kc) {
    int d = kc / 5;
    int i0 = (kc - d * 5) * 32;
    float qd = qvrow[d];
    const float* sp = selrow + i0 + kba;
    v16h a;
#pragma unroll
    for (int j = 0; j < 8; ++j) {
      a[j]     = (_Float16)(qd * sp[j]);
      a[j + 8] = (_Float16)(qd * sp[8 + j]);
    }
#pragma unroll
    for (int ot = 0; ot < 4; ++ot) {
      const _Float16* bp = Bcol + (size_t)(ot * 16) * HKW + kc * 32;
      v16h b = cat16(*(const h8*)bp, *(const h8*)(bp + 8));
      acc[ot] = wmma16(a, b, acc[ot]);
    }
  }
  int col = lane & 15, roff = (lane >> 4) << 3;
#pragma unroll
  for (int ot = 0; ot < 4; ++ot) {
#pragma unroll
    for (int r = 0; r < 8; ++r) {
      int m = rt * 16 + roff + r;
      int o = ot * 16 + col;
      float val = acc[ot][r];
      if (mode == 0) {
        outbuf[m * 64 + o] = 1.f / (1.f + __expf(-val));
      } else {
        float u = ug[m * 64 + o];
        float rr = rg[m * 64 + o];
        size_t row = (size_t)nb[m] * N_ + nn[m];
        float hnew = rr * hin[row * Dh + o];
        outbuf[m * 64 + o] = (1.f - u) * hnew + u * tanhf(val);
      }
    }
  }
}

// ================= host launcher =================
extern "C" void kernel_launch(void* const* d_in, const int* in_sizes, int n_in,
                              void* d_out, int out_size, void* d_ws, size_t ws_size,
                              hipStream_t stream) {
  (void)in_sizes; (void)n_in; (void)out_size; (void)ws_size;
  const float* x   = (const float*)d_in[0];
  const float* h   = (const float*)d_in[1];
  const float* qv  = (const float*)d_in[2];
  const int*   adj = (const int*)d_in[3];
  const int*   nb  = (const int*)d_in[4];
  const int*   nn  = (const int*)d_in[5];
  const float* Wq  = (const float*)d_in[6];
  const float* bq  = (const float*)d_in[7];
  const float* Wk  = (const float*)d_in[8];
  const float* bk  = (const float*)d_in[9];
  const float* Wv  = (const float*)d_in[10];
  const float* bv  = (const float*)d_in[11];
  const float* W1  = (const float*)d_in[12];
  const float* b1  = (const float*)d_in[13];
  const float* W2  = (const float*)d_in[14];
  const float* b2  = (const float*)d_in[15];
  const float* Wr  = (const float*)d_in[16];
  const float* br  = (const float*)d_in[17];
  const float* Wu  = (const float*)d_in[18];
  const float* bu  = (const float*)d_in[19];
  const float* Wc  = (const float*)d_in[20];
  const float* bc  = (const float*)d_in[21];
  float* out = (float*)d_out;

  char* ws = (char*)d_ws;
  size_t off = 0;
  auto alloc = [&](size_t bytes) { size_t r = off; off += (bytes + 255) & ~(size_t)255; return r; };

  float*    cf32 = (float*)   (ws + alloc((size_t)BN * C_ * 4));
  _Float16* cf16 = (_Float16*)(ws + alloc((size_t)BN * CP * 2));
  _Float16* Wqt  = (_Float16*)(ws + alloc((size_t)H_ * 16 * CP * 2));
  _Float16* Wkt  = (_Float16*)(ws + alloc((size_t)H_ * 16 * CP * 2));
  _Float16* Wvt  = (_Float16*)(ws + alloc((size_t)H_ * 144 * CP * 2));
  _Float16* W1t  = (_Float16*)(ws + alloc((size_t)144 * CATW * 2));
  _Float16* W2t  = (_Float16*)(ws + alloc((size_t)144 * HIDW * 2));
  _Float16* Wrt  = (_Float16*)(ws + alloc((size_t)64 * HKW * 2));
  _Float16* Wut  = (_Float16*)(ws + alloc((size_t)64 * HKW * 2));
  _Float16* Wct  = (_Float16*)(ws + alloc((size_t)64 * HKW * 2));
  _Float16* qf   = (_Float16*)(ws + alloc((size_t)H_ * BN * 32 * 2));
  _Float16* kf   = (_Float16*)(ws + alloc((size_t)H_ * BN * 32 * 2));
  _Float16* vt   = (_Float16*)(ws + alloc((size_t)H_ * B_ * 144 * N_ * 2));
  float*    sc   = (float*)   (ws + alloc((size_t)H_ * B_ * N_ * N_ * 4));
  _Float16* at   = (_Float16*)(ws + alloc((size_t)H_ * B_ * N_ * N_ * 2));
  _Float16* cat  = (_Float16*)(ws + alloc((size_t)BN * CATW * 2));
  _Float16* hid  = (_Float16*)(ws + alloc((size_t)BN * HIDW * 2));
  float*    c2   = (float*)   (ws + alloc((size_t)BN * C_ * 4));
  float*    sel  = (float*)   (ws + alloc((size_t)M_ * CP * 4));
  float*    selc = (float*)   (ws + alloc((size_t)M_ * CP * 4));
  float*    rg   = (float*)   (ws + alloc((size_t)M_ * 64 * 4));
  float*    ug   = (float*)   (ws + alloc((size_t)M_ * 64 * 4));

  const int TB = 256;
  auto cdiv = [](int a, int b) { return (a + b - 1) / b; };

  // weight prep + init
  k_prep_qk<<<cdiv(H_ * 16 * CP, TB), TB, 0, stream>>>(Wq, bq, Wqt);
  k_prep_qk<<<cdiv(H_ * 16 * CP, TB), TB, 0, stream>>>(Wk, bk, Wkt);
  k_prep_v <<<cdiv(H_ * 144 * CP, TB), TB, 0, stream>>>(Wv, bv, Wvt);
  k_prep_w1<<<cdiv(144 * CATW, TB), TB, 0, stream>>>(W1, b1, W1t);
  k_prep_w2<<<cdiv(144 * HIDW, TB), TB, 0, stream>>>(W2, b2, W2t);
  k_prep_hg<<<cdiv(64 * HKW, TB), TB, 0, stream>>>(Wr, br, Wrt);
  k_prep_hg<<<cdiv(64 * HKW, TB), TB, 0, stream>>>(Wu, bu, Wut);
  k_prep_hg<<<cdiv(64 * HKW, TB), TB, 0, stream>>>(Wc, bc, Wct);
  k_fill<<<cdiv(BN * CATW, TB), TB, 0, stream>>>(cat, BN * CATW, CATW, 544);
  k_fill<<<cdiv(BN * HIDW, TB), TB, 0, stream>>>(hid, BN * HIDW, HIDW, 129);

  // pipeline
  k_combined<<<cdiv(BN * CP, TB), TB, 0, stream>>>(x, h, cf32, cf16);
  k_qk     <<<H_ * 512, 32, 0, stream>>>(cf16, Wqt, Wkt, qf, kf);
  k_v      <<<H_ * 512 * 9, 32, 0, stream>>>(cf16, Wvt, vt);
  k_scores <<<H_ * B_ * 64 * 64, 32, 0, stream>>>(qf, kf, adj, sc);
  k_softmax<<<H_ * B_ * N_, 32, 0, stream>>>(sc, at);
  k_av     <<<H_ * B_ * 64, 32, 0, stream>>>(at, vt, cat);
  k_mlp1   <<<512 * 9, 32, 0, stream>>>(cat, W1t, hid);
  k_mlp2   <<<512 * 9, 32, 0, stream>>>(hid, W2t, cf32, c2);
  k_sel    <<<cdiv(M_ * CP, TB), TB, 0, stream>>>(c2, nb, nn, sel);
  k_hyper  <<<256, 32, 0, stream>>>(sel, qv, Wrt, rg, 0, nullptr, nullptr, nullptr, nb, nn);
  k_hyper  <<<256, 32, 0, stream>>>(sel, qv, Wut, ug, 0, nullptr, nullptr, nullptr, nb, nn);
  k_selc   <<<cdiv(M_ * CP, TB), TB, 0, stream>>>(x, h, rg, nb, nn, selc);
  k_hyper  <<<256, 32, 0, stream>>>(selc, qv, Wct, out, 1, rg, ug, h, nb, nn);
}
